// GCNLayer_11965778887252
// MI455X (gfx1250) — matrix operand order repaired and checked by reading, not compiled, not run on hardware
//
#include <hip/hip_runtime.h>
#include <hip/hip_bf16.h>

typedef __attribute__((ext_vector_type(16))) _Float16 v16h;
typedef __attribute__((ext_vector_type(8)))  float    v8f;
typedef __attribute__((ext_vector_type(4)))  float    v4f;

#define N_NODES 16384
#define FDIM    64
#define KSPLIT  4
#define KCHUNK  (N_NODES / KSPLIT)

// ---------------------------------------------------------------------------
// Kernel 1: deg_i = sum_j A[i][j]; dis[i] = deg>0 ? rsqrt(deg) : 0
// 1 block (256 threads) per row, float4-coalesced stream of the row.
// ---------------------------------------------------------------------------
__global__ __launch_bounds__(256) void gcn_deg(const float* __restrict__ A,
                                               float* __restrict__ dis) {
    __shared__ float red[256];
    const v4f* Av = (const v4f*)(A + (size_t)blockIdx.x * N_NODES);
    float s = 0.f;
    for (int c = threadIdx.x; c < N_NODES / 4; c += 256) {
        v4f v = Av[c];
        s += v.x + v.y + v.z + v.w;
    }
    red[threadIdx.x] = s;
    __syncthreads();
    for (int off = 128; off > 0; off >>= 1) {
        if ((int)threadIdx.x < off) red[threadIdx.x] += red[threadIdx.x + off];
        __syncthreads();
    }
    if (threadIdx.x == 0) {
        float d = red[0];
        dis[blockIdx.x] = d > 0.f ? rsqrtf(d) : 0.f;
    }
}

// ---------------------------------------------------------------------------
// Kernel 2: Zt[f][j] = (f16)( dis[j] * sum_k x[j][k] * W[k][f] )
// Stored transposed (64 x 16384) so the WMMA B-fragment loads are contiguous.
// One block of 256 threads handles 4 rows j (64 f-lanes x 4 rows).
// ---------------------------------------------------------------------------
__global__ __launch_bounds__(256) void gcn_xw(const float* __restrict__ x,
                                              const float* __restrict__ W,
                                              const float* __restrict__ dis,
                                              _Float16* __restrict__ Zt) {
    __shared__ float sW[FDIM * FDIM];   // 16 KiB
    __shared__ float sx[4 * FDIM];
    for (int i = threadIdx.x; i < FDIM * FDIM; i += 256) sW[i] = W[i];
    const int j0 = blockIdx.x * 4;
    sx[threadIdx.x] = x[(size_t)j0 * FDIM + threadIdx.x];
    __syncthreads();

    const int f  = threadIdx.x & 63;
    const int jj = threadIdx.x >> 6;
    float s = 0.f;
#pragma unroll
    for (int k = 0; k < FDIM; ++k) s += sx[jj * FDIM + k] * sW[k * FDIM + f];
    const int j = j0 + jj;
    Zt[(size_t)f * N_NODES + j] = (_Float16)(dis[j] * s);
}

// ---------------------------------------------------------------------------
// Kernel 3: partial[ks][i][f] = sum_{k in chunk ks} A[i][k] * Zt[f][k]
// Wave-level: 16-row M-tile x 64 cols (4 WMMA accumulators), f16 inputs,
// fp32 accumulation via v_wmma_f32_16x16x32_f16.
// grid = (N/64 M-blocks, KSPLIT), block = 128 threads = 4 waves.
// ---------------------------------------------------------------------------
__global__ __launch_bounds__(128) void gcn_spmm(const float* __restrict__ A,
                                                const _Float16* __restrict__ Zt,
                                                float* __restrict__ part) {
    const int lane = threadIdx.x & 31;
    const int wave = threadIdx.x >> 5;
    const int mrow = lane & 15;   // A-fragment row / B-fragment col / C col
    const int hgrp = lane >> 4;   // half-wave group

    const long row0  = (long)blockIdx.x * 64 + (long)wave * 16;
    const long kbase = (long)blockIdx.y * KCHUNK;

    // A fragment: elems 0..15 -> K = h*16 + 0..15
    const float*    Arow  = A  + (row0 + mrow) * (long)N_NODES + kbase + hgrp * 16;
    // B fragment: lane holds col n = lane&15, 16 contiguous K halfs at h*16
    const _Float16* Zlane = Zt + (long)mrow * N_NODES + kbase + hgrp * 16;

    v8f acc0 = {}, acc1 = {}, acc2 = {}, acc3 = {};

    for (int kk = 0; kk < KCHUNK; kk += 32) {
        v4f a0 = *(const v4f*)(Arow + kk);
        v4f a1 = *(const v4f*)(Arow + kk + 4);
        v4f a2 = *(const v4f*)(Arow + kk + 8);
        v4f a3 = *(const v4f*)(Arow + kk + 12);
        v16h a;
#pragma unroll
        for (int e = 0; e < 4; ++e) {
            a[e]      = (_Float16)a0[e];
            a[e + 4]  = (_Float16)a1[e];
            a[e + 8]  = (_Float16)a2[e];
            a[e + 12] = (_Float16)a3[e];
        }
        v16h b0 = *(const v16h*)(Zlane + kk);
        v16h b1 = *(const v16h*)(Zlane + 16L * N_NODES + kk);
        v16h b2 = *(const v16h*)(Zlane + 32L * N_NODES + kk);
        v16h b3 = *(const v16h*)(Zlane + 48L * N_NODES + kk);

        acc0 = __builtin_amdgcn_wmma_f32_16x16x32_f16(false, a, false, b0, (short)0, acc0, false, false);
        acc1 = __builtin_amdgcn_wmma_f32_16x16x32_f16(false, a, false, b1, (short)0, acc1, false, false);
        acc2 = __builtin_amdgcn_wmma_f32_16x16x32_f16(false, a, false, b2, (short)0, acc2, false, false);
        acc3 = __builtin_amdgcn_wmma_f32_16x16x32_f16(false, a, false, b3, (short)0, acc3, false, false);
    }

    // C/D layout: VGPR r -> row (h*8 + r), col = lane&15
    float* p = part + ((long)blockIdx.y * N_NODES + row0) * FDIM;
#pragma unroll
    for (int r = 0; r < 8; ++r) {
        const long rr = (long)(hgrp * 8 + r) * FDIM;
        p[rr +      mrow] = acc0[r];
        p[rr + 16 + mrow] = acc1[r];
        p[rr + 32 + mrow] = acc2[r];
        p[rr + 48 + mrow] = acc3[r];
    }
}

// ---------------------------------------------------------------------------
// Kernel 4: out[i][f] = dis[i] * sum_ks partial[ks][i][f]   (deterministic)
// ---------------------------------------------------------------------------
__global__ __launch_bounds__(256) void gcn_reduce(const float* __restrict__ part,
                                                  const float* __restrict__ dis,
                                                  float* __restrict__ out) {
    const long idx = (long)blockIdx.x * 256 + threadIdx.x;  // i*64 + f
    float s = 0.f;
#pragma unroll
    for (int ks = 0; ks < KSPLIT; ++ks)
        s += part[(long)ks * N_NODES * FDIM + idx];
    out[idx] = dis[idx >> 6] * s;
}

// ---------------------------------------------------------------------------
extern "C" void kernel_launch(void* const* d_in, const int* in_sizes, int n_in,
                              void* d_out, int out_size, void* d_ws, size_t ws_size,
                              hipStream_t stream) {
    const float* x = (const float*)d_in[0];   // (N, 64)
    const float* A = (const float*)d_in[1];   // (N, N)
    const float* W = (const float*)d_in[2];   // (64, 64)
    float* out = (float*)d_out;

    char* ws = (char*)d_ws;
    float*    dis  = (float*)ws;                                   //  64 KiB
    _Float16* Zt   = (_Float16*)(ws + (64 << 10));                 //   2 MiB
    float*    part = (float*)(ws + (64 << 10) + (2 << 20));        //  16 MiB (KSPLIT * N * 64 * 4)

    gcn_deg   <<<N_NODES,                 256, 0, stream>>>(A, dis);
    gcn_xw    <<<N_NODES / 4,             256, 0, stream>>>(x, W, dis, Zt);
    gcn_spmm  <<<dim3(N_NODES / 64, KSPLIT), 128, 0, stream>>>(A, Zt, part);
    gcn_reduce<<<(N_NODES * FDIM) / 256,  256, 0, stream>>>(part, dis, out);
}
